// gcn_15109694947748
// MI455X (gfx1250) — matrix operand order repaired and checked by reading, not compiled, not run on hardware
//
#include <hip/hip_runtime.h>
#include <hip/hip_bf16.h>
#include <stdint.h>

// ---------------------------------------------------------------------------
// MI455X (gfx1250) implementation. Heavy GEMM-shaped stages use
// v_wmma_f32_16x16x32_f16 (wave32, one wave per workgroup, EXEC all ones),
// batched so WMMAs issue back-to-back after a single load clause.
// zconv stages its LDS tile with GLOBAL_LOAD_ASYNC_TO_LDS_B128 (ASYNCcnt).
// Glue stages (BN stats/finalize, scatter, pooling, LSTM cell, attention)
// are plain VALU kernels.
// ---------------------------------------------------------------------------

typedef __attribute__((ext_vector_type(16))) _Float16 v16h;
typedef __attribute__((ext_vector_type(8)))  float    v8f;

#define WMMA16(acc, A, B) \
  __builtin_amdgcn_wmma_f32_16x16x32_f16(false, (A), false, (B), (short)0, (acc), false, false)

// problem constants
constexpr int BATCH = 64;        // B0
constexpr int NCH   = 59;
constexpr int TLEN  = 300;
constexpr int WIN   = 5;
constexpr int STEP  = 50;
constexpr int N320  = BATCH * WIN;        // 320
constexpr int NNODE = BATCH * NCH;        // 3776
constexpr int MROWS = NNODE * WIN;        // 18880
constexpr int EDGES = 216 * BATCH;        // 13824
constexpr float CNT1  = 1888000.f;        // 320*59*100
constexpr float CNT13 = 2832000.f;        // 320*59*150

// stats buffer layout (floats)
constexpr int S_SUM1  = 0,   S_SQ1  = 32, S_SC1  = 64,  S_SH1  = 96;
constexpr int S_SUM13 = 128, S_SQ13 = 144, S_SC13 = 160, S_SH13 = 176;

__device__ __forceinline__ int a_koff(int h) {
  return h;
}

// ------------------------- generic fill ------------------------------------
__global__ void fill_kernel(float* __restrict__ p, int n) {
  int i = blockIdx.x * blockDim.x + threadIdx.x;
  if (i < n) p[i] = 0.f;
}

// ------------------------- conv blocks (WMMA) ------------------------------
// One wave per (n320, ch).  Y(20x100) = W(20xKK) x Patch(KKx100).
// Pass EMIT=false: accumulate per-oc sum/sumsq of raw conv (bias cancels in BN).
// Pass EMIT=true : BN+leaky, pool2 -> p(20x50), v[o][u]=sum_c p*w13[o][c],
//                  store v (f16) and accumulate BN13 stats.
template<int KK, int KT, int PAD, bool EMIT>
__global__ __launch_bounds__(32)
void conv_kernel(const float* __restrict__ x, const float* __restrict__ w,
                 float* __restrict__ stats, const float* __restrict__ w13,
                 _Float16* __restrict__ vout, int wofs)
{
  constexpr int KPAD = KT * 32;
  constexpr int LEN  = 112 + KPAD;
  __shared__ _Float16 xpad[LEN];
  __shared__ float ybuf[20 * 112];
  __shared__ float ls[32], lq[32];

  const int L  = threadIdx.x;
  const int wg = blockIdx.x;              // 18880
  const int n  = wg / NCH, ch = wg % NCH;
  const int b  = n / WIN,  w0 = n % WIN;
  const float* xrow = x + (size_t)(b * NCH + ch) * TLEN + w0 * STEP;
  __builtin_prefetch(xrow, 0, 1);

  for (int q = L; q < LEN; q += 32) {
    int t = q - PAD;
    xpad[q] = (t >= 0 && t < 100) ? (_Float16)xrow[t] : (_Float16)0.f;
  }
  ls[L] = 0.f; lq[L] = 0.f;
  __syncthreads();

  const int lm   = L & 15;
  const int kbA  = (L < 16) ? 0 : 16;
  const int kbB  = (L < 16) ? 0 : 16;
  const int mofs = (L < 16) ? 0 : 8;

  // hoisted A fragments (weights)
  v16h a[2][KT];
#pragma unroll
  for (int mt = 0; mt < 2; ++mt) {
    int row = mt * 16 + lm;
#pragma unroll
    for (int kt = 0; kt < KT; ++kt) {
      v16h av;
#pragma unroll
      for (int h = 0; h < 16; ++h) {
        int k = kt * 32 + kbA + a_koff(h);
        float val = (row < 20 && k < KK) ? w[row * KK + k] : 0.f;
        av[h] = (_Float16)val;
      }
      a[mt][kt] = av;
    }
  }

#pragma unroll 1
  for (int nt = 0; nt < 7; ++nt) {
    const int tcol = nt * 16 + lm;
    // batch all B fragments for this n-tile, then burst the WMMAs
    v16h bvs[KT];
#pragma unroll
    for (int kt = 0; kt < KT; ++kt) {
      v16h bv;
      int base = tcol + kt * 32 + kbB;
#pragma unroll
      for (int h = 0; h < 16; ++h) bv[h] = xpad[base + h];
      bvs[kt] = bv;
    }
    v8f acc0 = {}; v8f acc1 = {};
#pragma unroll
    for (int kt = 0; kt < KT; ++kt) {
      acc0 = WMMA16(acc0, a[0][kt], bvs[kt]);
      acc1 = WMMA16(acc1, a[1][kt], bvs[kt]);
    }
    if (tcol < 100) {
#pragma unroll
      for (int r = 0; r < 8; ++r) {
        int oc0 = r + mofs;          // < 16
        int oc1 = 16 + r + mofs;     // valid only if < 20
        float y0 = acc0[r], y1 = acc1[r];
        if (!EMIT) {
          atomicAdd(&ls[oc0], y0); atomicAdd(&lq[oc0], y0 * y0);
          if (oc1 < 20) { atomicAdd(&ls[oc1], y1); atomicAdd(&lq[oc1], y1 * y1); }
        } else {
          float sc = stats[S_SC1 + oc0], sh = stats[S_SH1 + oc0];
          float yv = sc * y0 + sh; yv = yv > 0.f ? yv : 0.01f * yv;
          ybuf[oc0 * 112 + tcol] = yv;
          if (oc1 < 20) {
            sc = stats[S_SC1 + oc1]; sh = stats[S_SH1 + oc1];
            yv = sc * y1 + sh; yv = yv > 0.f ? yv : 0.01f * yv;
            ybuf[oc1 * 112 + tcol] = yv;
          }
        }
      }
    }
  }
  __syncthreads();

  if (!EMIT) {
    if (L < 20) {
      atomicAdd(&stats[S_SUM1 + L], ls[L]);
      atomicAdd(&stats[S_SQ1  + L], lq[L]);
    }
  } else {
    // pool2 + 1x1 conv (w13) + BN13 stats
    for (int idx = L; idx < 500; idx += 32) {
      int o = idx / 50, u = idx % 50;
      float acc = 0.f;
#pragma unroll
      for (int c = 0; c < 20; ++c) {
        float p = 0.5f * (ybuf[c * 112 + 2 * u] + ybuf[c * 112 + 2 * u + 1]);
        acc += p * w13[o * 20 + c];
      }
      vout[((size_t)(n * 10 + o) * NCH + ch) * 150 + wofs + u] = (_Float16)acc;
      atomicAdd(&ls[o], acc); atomicAdd(&lq[o], acc * acc);
    }
    __syncthreads();
    if (L < 10) {
      atomicAdd(&stats[S_SUM13 + L], ls[L]);
      atomicAdd(&stats[S_SQ13  + L], lq[L]);
    }
  }
}

// BN finalize: scale = g*rsqrt(var+eps), shift = be - mu*scale
__global__ void fin_kernel(float* __restrict__ stats, const float* __restrict__ g,
                           const float* __restrict__ be, int nch, float cnt,
                           int so, int qo, int co, int ho)
{
  int i = threadIdx.x;
  if (i < nch) {
    float mu  = stats[so + i] / cnt;
    float var = stats[qo + i] / cnt - mu * mu;
    float sc  = g[i] * rsqrtf(var + 1e-5f);
    stats[co + i] = sc;
    stats[ho + i] = be[i] - mu * sc;
  }
}

// feat16[m][k] (padded K=160): sum over o of leaky(BN13(v))
__global__ void feat_kernel(const _Float16* __restrict__ v16,
                            const float* __restrict__ stats,
                            _Float16* __restrict__ feat16)
{
  int idx = blockIdx.x * blockDim.x + threadIdx.x;
  if (idx >= MROWS * 160) return;
  int m = idx / 160, k = idx % 160;
  float val = 0.f;
  if (k < 150) {
    int node = m / WIN, w0 = m % WIN;
    int b = node / NCH, ch = node % NCH;
    int n = b * WIN + w0;
#pragma unroll
    for (int o = 0; o < 10; ++o) {
      float v = (float)v16[((size_t)(n * 10 + o) * NCH + ch) * 150 + k];
      v = stats[S_SC13 + o] * v + stats[S_SH13 + o];
      val += (v > 0.f ? v : 0.01f * v);
    }
  }
  feat16[idx] = (_Float16)val;
}

// gcn_w (150,100) -> gwT[nn][k] f16, padded to 112x160, B-fragment friendly
__global__ void gwprep_kernel(const float* __restrict__ gcn_w, _Float16* __restrict__ gwT)
{
  int idx = blockIdx.x * blockDim.x + threadIdx.x;
  if (idx >= 112 * 160) return;
  int nn = idx / 160, k = idx % 160;
  gwT[idx] = (nn < 100 && k < 150) ? (_Float16)gcn_w[k * 100 + nn] : (_Float16)0.f;
}

// h(18880x100) = feat16(18880x150) @ gcn_w(150x100), WMMA
__global__ __launch_bounds__(32)
void gcn_mm_kernel(const _Float16* __restrict__ feat16,
                   const _Float16* __restrict__ gwT, float* __restrict__ hout)
{
  const int L = threadIdx.x, mt = blockIdx.x;   // 1180 mtiles
  const int lm = L & 15;
  const int row = mt * 16 + lm;
  const int kbA = (L < 16) ? 0 : 16, kbB = (L < 16) ? 0 : 16, mofs = (L < 16) ? 0 : 8;

  __builtin_prefetch(feat16 + (size_t)row * 160, 0, 1);

  // hoist all 5 A fragments (K = 160)
  v16h afr[5];
#pragma unroll
  for (int kt = 0; kt < 5; ++kt) {
    v16h av;
    const _Float16* fr = feat16 + (size_t)row * 160 + kt * 32 + kbA;
#pragma unroll
    for (int h = 0; h < 16; ++h) av[h] = fr[a_koff(h)];
    afr[kt] = av;
  }

#pragma unroll 1
  for (int nt = 0; nt < 7; ++nt) {
    const int nn = nt * 16 + lm;
    // batch all B fragments (contiguous 32B chunks), then burst the WMMAs
    v16h bvs[5];
#pragma unroll
    for (int kt = 0; kt < 5; ++kt) {
      v16h bv;
      const _Float16* br = gwT + (size_t)nn * 160 + kt * 32 + kbB;
#pragma unroll
      for (int h = 0; h < 16; ++h) bv[h] = br[h];
      bvs[kt] = bv;
    }
    v8f acc = {};
#pragma unroll
    for (int kt = 0; kt < 5; ++kt) acc = WMMA16(acc, afr[kt], bvs[kt]);
    if (nn < 100) {
#pragma unroll
      for (int r = 0; r < 8; ++r) {
        int m = mt * 16 + r + mofs;
        hout[(size_t)m * 100 + nn] = acc[r];
      }
    }
  }
}

// ------------------------- GCN scatter / BN2 -------------------------------
__global__ void deg_kernel(const int* __restrict__ ei, const float* __restrict__ ew,
                           float* __restrict__ deg)
{
  int e = blockIdx.x * blockDim.x + threadIdx.x;
  if (e >= EDGES) return;
  int col = ei[EDGES + e];
  float w = ew[e % 216]; w = w > 0.f ? w : 0.f;
  atomicAdd(&deg[col], w);
}

__global__ void dinv_kernel(const float* __restrict__ deg, float* __restrict__ dinv)
{
  int i = blockIdx.x * blockDim.x + threadIdx.x;
  if (i < NNODE) dinv[i] = rsqrtf(deg[i] + 1.f);
}

__global__ void agginit_kernel(const float* __restrict__ h, const float* __restrict__ dinv,
                               float* __restrict__ agg)
{
  int idx = blockIdx.x * blockDim.x + threadIdx.x;
  if (idx >= MROWS * 100) return;
  int node = (idx / 100) / WIN;
  float d = dinv[node];
  agg[idx] = h[idx] * d * d;      // gcn_b cancels in BN2
}

__global__ __launch_bounds__(128)
void scatter_kernel(const int* __restrict__ ei, const float* __restrict__ ew,
                    const float* __restrict__ dinv, const float* __restrict__ h,
                    float* __restrict__ agg)
{
  int e = blockIdx.x;
  int row = ei[e], col = ei[EDGES + e];
  float w = ew[e % 216]; w = w > 0.f ? w : 0.f;
  float nrm = dinv[row] * w * dinv[col];
  for (int i = threadIdx.x; i < 500; i += 128)
    atomicAdd(&agg[(size_t)col * 500 + i], h[(size_t)row * 500 + i] * nrm);
}

__global__ __launch_bounds__(256)
void bn2_kernel(const float* __restrict__ agg, const float* __restrict__ g,
                const float* __restrict__ be, float* __restrict__ sc2,
                float* __restrict__ sh2)
{
  __shared__ float rs[256], rq[256];
  int wf = blockIdx.x;                 // (win,f) pair, 500 of them
  float s = 0.f, q = 0.f;
  for (int node = threadIdx.x; node < NNODE; node += 256) {
    float v = agg[(size_t)node * 500 + wf];
    s += v; q += v * v;
  }
  rs[threadIdx.x] = s; rq[threadIdx.x] = q; __syncthreads();
  for (int st = 128; st > 0; st >>= 1) {
    if (threadIdx.x < st) { rs[threadIdx.x] += rs[threadIdx.x + st]; rq[threadIdx.x] += rq[threadIdx.x + st]; }
    __syncthreads();
  }
  if (threadIdx.x == 0) {
    float mu = rs[0] / (float)NNODE, var = rq[0] / (float)NNODE - mu * mu;
    int f = wf % 100;
    float sc = g[f] * rsqrtf(var + 1e-5f);
    sc2[wf] = sc; sh2[wf] = be[f] - mu * sc;
  }
}

__global__ void gout_kernel(float* __restrict__ agg, const float* __restrict__ sc2,
                            const float* __restrict__ sh2)
{
  int idx = blockIdx.x * blockDim.x + threadIdx.x;
  if (idx >= MROWS * 100) return;
  int wf = ((idx / 100) % WIN) * 100 + (idx % 100);
  agg[idx] = agg[idx] * sc2[wf] + sh2[wf];
}

// ------------------------- conv11 + pool -> seq ----------------------------
// LDS tile staged via GLOBAL_LOAD_ASYNC_TO_LDS_B128 (ASYNCcnt path).
__global__ __launch_bounds__(128)
void zconv_kernel(const float* __restrict__ gout, const float* __restrict__ w11,
                  const float* __restrict__ b11, float* __restrict__ seq)
{
  __shared__ __align__(16) float zs[59 * 100];
  __shared__ float ys[10 * 100];
  __shared__ float ws[10 * 59 * 2];
  int tid = threadIdx.x;
  int nb = blockIdx.x;                 // 320
  int b = nb / WIN, w0 = nb % WIN;

  // async copy of the 59x100 f32 tile: row ch lives at ((b*59+ch)*5+w0)*400 B,
  // 25 x b128 chunks per row.  LDS operand = workgroup-relative offset
  // (low 32 bits of the generic shared pointer).
  {
    uint64_t gbase = (uint64_t)(uintptr_t)gout + ((uint64_t)(b * NCH) * WIN + w0) * 400u;
    uint32_t lbase = (uint32_t)(uintptr_t)(void*)zs;
    for (int i = tid; i < 59 * 25; i += 128) {
      int ch = i / 25, c16 = i % 25;
      uint64_t ga = gbase + (uint64_t)ch * 2000u + (uint64_t)c16 * 16u;
      uint32_t la = lbase + (uint32_t)(ch * 400 + c16 * 16);
      asm volatile("global_load_async_to_lds_b128 %0, %1, off"
                   :: "v"(la), "v"(ga) : "memory");
    }
    asm volatile("s_wait_asynccnt 0" ::: "memory");
  }
  for (int i = tid; i < 1180; i += 128) ws[i] = w11[i];
  __syncthreads();

  for (int i = tid; i < 10 * 99; i += 128) {
    int o = i / 99, t = i % 99;
    float acc = b11[o];
    for (int ch = 0; ch < 59; ++ch) {
      acc += zs[ch * 100 + t]     * ws[(o * 59 + ch) * 2 + 0];
      acc += zs[ch * 100 + t + 1] * ws[(o * 59 + ch) * 2 + 1];
    }
    ys[o * 100 + t] = acc > 0.f ? acc : 0.01f * acc;
  }
  __syncthreads();
  for (int i = tid; i < 490; i += 128) {
    int o = i / 49, u = i % 49;
    seq[(size_t)(b * WIN + w0) * 490 + o * 49 + u] =
        0.5f * (ys[o * 100 + 2 * u] + ys[o * 100 + 2 * u + 1]);
  }
}

// ------------------------- LSTM gates (WMMA) -------------------------------
// G(64x1024) = X(64xKx) @ Wx^T + H(64x256) @ Wh^T
__global__ __launch_bounds__(32)
void lstm_gates_kernel(const float* __restrict__ X, int xstride, int Kx, int KTx,
                       const float* __restrict__ Wx, const float* __restrict__ H,
                       const float* __restrict__ Wh, float* __restrict__ G)
{
  const int L = threadIdx.x;
  const int wg = blockIdx.x;           // 256 = 4 mtiles * 64 ntiles
  const int mt = wg >> 6, ntl = wg & 63;
  const int lm = L & 15;
  const int m = mt * 16 + lm;
  const int j = ntl * 16 + lm;
  const int kbA = (L < 16) ? 0 : 16, kbB = (L < 16) ? 0 : 16, mofs = (L < 16) ? 0 : 8;

  __builtin_prefetch(Wx + (size_t)j * Kx, 0, 1);
  __builtin_prefetch(Wh + (size_t)j * 256, 0, 1);

  v8f acc = {};
#pragma unroll 1
  for (int kt = 0; kt < KTx; ++kt) {
    v16h av, bv;
#pragma unroll
    for (int h = 0; h < 16; ++h) {
      int k = kt * 32 + kbA + a_koff(h);
      av[h] = (k < Kx) ? (_Float16)X[(size_t)m * xstride + k] : (_Float16)0.f;
    }
#pragma unroll
    for (int h = 0; h < 16; ++h) {
      int k = kt * 32 + kbB + h;
      bv[h] = (k < Kx) ? (_Float16)Wx[(size_t)j * Kx + k] : (_Float16)0.f;
    }
    acc = WMMA16(acc, av, bv);
  }
#pragma unroll 1
  for (int kt = 0; kt < 8; ++kt) {     // K = 256 hidden
    v16h av, bv;
#pragma unroll
    for (int h = 0; h < 16; ++h)
      av[h] = (_Float16)H[(size_t)m * 256 + kt * 32 + kbA + a_koff(h)];
#pragma unroll
    for (int h = 0; h < 16; ++h)
      bv[h] = (_Float16)Wh[(size_t)j * 256 + kt * 32 + kbB + h];
    acc = WMMA16(acc, av, bv);
  }
#pragma unroll
  for (int r = 0; r < 8; ++r) {
    int bm = mt * 16 + r + mofs;
    G[(size_t)bm * 1024 + j] = acc[r];
  }
}

__global__ __launch_bounds__(256)
void lstm_cell_kernel(const float* __restrict__ G, const float* __restrict__ bih,
                      const float* __restrict__ bhh, float* __restrict__ hbuf,
                      float* __restrict__ cbuf, float* __restrict__ outp, int ostride)
{
  int idx = blockIdx.x * blockDim.x + threadIdx.x;
  if (idx >= BATCH * 256) return;
  int b = idx >> 8, hh = idx & 255;
  const float* g = G + (size_t)b * 1024;
  float gi = g[hh]        + bih[hh]        + bhh[hh];
  float gf = g[256 + hh]  + bih[256 + hh]  + bhh[256 + hh];
  float gg = g[512 + hh]  + bih[512 + hh]  + bhh[512 + hh];
  float go = g[768 + hh]  + bih[768 + hh]  + bhh[768 + hh];
  float c  = cbuf[idx];
  float si = 1.f / (1.f + expf(-gi));
  float sf = 1.f / (1.f + expf(-gf));
  float so = 1.f / (1.f + expf(-go));
  c = sf * c + si * tanhf(gg);
  float hv = so * tanhf(c);
  cbuf[idx] = c; hbuf[idx] = hv;
  outp[(size_t)b * ostride + hh] = hv;
}

// ------------------------- attention ---------------------------------------
__global__ __launch_bounds__(256)
void attn_kernel(const float* __restrict__ out2, const float* __restrict__ hlast,
                 float* __restrict__ out)
{
  __shared__ float red[256];
  __shared__ float at[WIN];
  int b = blockIdx.x, tid = threadIdx.x;
  float hl = hlast[(size_t)b * 256 + tid];
  for (int t = 0; t < WIN; ++t) {
    red[tid] = out2[((size_t)b * WIN + t) * 256 + tid] * hl;
    __syncthreads();
    for (int st = 128; st > 0; st >>= 1) {
      if (tid < st) red[tid] += red[tid + st];
      __syncthreads();
    }
    if (tid == 0) at[t] = red[0];
    __syncthreads();
  }
  if (tid == 0) {
    float mx = at[0];
    for (int t = 1; t < WIN; ++t) mx = at[t] > mx ? at[t] : mx;
    float s = 0.f;
    for (int t = 0; t < WIN; ++t) { at[t] = expf(at[t] - mx); s += at[t]; }
    for (int t = 0; t < WIN; ++t) at[t] /= s;
  }
  __syncthreads();
  float acc = 0.f;
  for (int t = 0; t < WIN; ++t)
    acc += out2[((size_t)b * WIN + t) * 256 + tid] * at[t];
  out[(size_t)b * 256 + tid] = acc;
}

// ---------------------------------------------------------------------------
extern "C" void kernel_launch(void* const* d_in, const int* in_sizes, int n_in,
                              void* d_out, int out_size, void* d_ws, size_t ws_size,
                              hipStream_t stream)
{
  const float* x      = (const float*)d_in[0];
  const float* w1     = (const float*)d_in[1];
  const float* g1     = (const float*)d_in[3];
  const float* be1    = (const float*)d_in[4];
  const float* w2     = (const float*)d_in[5];
  const float* g2     = (const float*)d_in[7];
  const float* be2    = (const float*)d_in[8];
  const float* w3     = (const float*)d_in[9];
  const float* g3     = (const float*)d_in[11];
  const float* be3    = (const float*)d_in[12];
  const float* w13    = (const float*)d_in[13];
  const float* g13    = (const float*)d_in[15];
  const float* be13   = (const float*)d_in[16];
  const float* gcn_w  = (const float*)d_in[17];
  const float* edge_w = (const float*)d_in[19];
  const float* bn2_g  = (const float*)d_in[20];
  const float* bn2_b  = (const float*)d_in[21];
  const float* w11    = (const float*)d_in[22];
  const float* b11    = (const float*)d_in[23];
  const float* wih0   = (const float*)d_in[24];
  const float* whh0   = (const float*)d_in[25];
  const float* bih0   = (const float*)d_in[26];
  const float* bhh0   = (const float*)d_in[27];
  const float* wih1   = (const float*)d_in[28];
  const float* whh1   = (const float*)d_in[29];
  const float* bih1   = (const float*)d_in[30];
  const float* bhh1   = (const float*)d_in[31];
  const int*   eidx   = (const int*)d_in[32];
  float* out = (float*)d_out;

  // workspace carve
  char* base = (char*)d_ws;
  size_t off = 0;
  auto alloc = [&](size_t bytes) -> void* {
    void* p = base + off;
    off += (bytes + 255) & ~(size_t)255;
    return p;
  };
  float*     stats  = (float*)alloc(256 * 4);
  _Float16*  v16    = (_Float16*)alloc((size_t)N320 * 10 * NCH * 150 * 2);
  _Float16*  feat16 = (_Float16*)alloc((size_t)MROWS * 160 * 2);
  _Float16*  gwT    = (_Float16*)alloc((size_t)112 * 160 * 2);
  float*     hbuf   = (float*)alloc((size_t)MROWS * 100 * 4);
  float*     agg    = (float*)alloc((size_t)MROWS * 100 * 4);
  float*     deg    = (float*)alloc(NNODE * 4);
  float*     dinv   = (float*)alloc(NNODE * 4);
  float*     sc2    = (float*)alloc(500 * 4);
  float*     sh2    = (float*)alloc(500 * 4);
  float*     seq    = (float*)alloc((size_t)BATCH * WIN * 490 * 4);
  float*     gates  = (float*)alloc((size_t)BATCH * 1024 * 4);
  float*     h0     = (float*)alloc((size_t)BATCH * 256 * 4);
  float*     c0     = (float*)alloc((size_t)BATCH * 256 * 4);
  float*     h1     = (float*)alloc((size_t)BATCH * 256 * 4);
  float*     c1     = (float*)alloc((size_t)BATCH * 256 * 4);
  float*     out1   = (float*)alloc((size_t)BATCH * WIN * 256 * 4);
  float*     out2   = (float*)alloc((size_t)BATCH * WIN * 256 * 4);

  const int NWG = N320 * NCH;   // 18880 conv workgroups

  // zero BN13 accumulators
  fill_kernel<<<1, 256, 0, stream>>>(stats + S_SUM13, 32);

  // ---- conv block 1: K=125, pad=62, wofs 0 ----
  fill_kernel<<<1, 64, 0, stream>>>(stats + S_SUM1, 64);
  conv_kernel<125, 4, 62, false><<<NWG, 32, 0, stream>>>(x, w1, stats, nullptr, nullptr, 0);
  fin_kernel<<<1, 32, 0, stream>>>(stats, g1, be1, 20, CNT1, S_SUM1, S_SQ1, S_SC1, S_SH1);
  conv_kernel<125, 4, 62, true><<<NWG, 32, 0, stream>>>(x, w1, stats, w13, v16, 0);

  // ---- conv block 2: K=59, pad=29, wofs 50 ----
  fill_kernel<<<1, 64, 0, stream>>>(stats + S_SUM1, 64);
  conv_kernel<59, 2, 29, false><<<NWG, 32, 0, stream>>>(x, w2, stats, nullptr, nullptr, 0);
  fin_kernel<<<1, 32, 0, stream>>>(stats, g2, be2, 20, CNT1, S_SUM1, S_SQ1, S_SC1, S_SH1);
  conv_kernel<59, 2, 29, true><<<NWG, 32, 0, stream>>>(x, w2, stats, w13, v16, 50);

  // ---- conv block 3: K=31, pad=15, wofs 100 ----
  fill_kernel<<<1, 64, 0, stream>>>(stats + S_SUM1, 64);
  conv_kernel<31, 1, 15, false><<<NWG, 32, 0, stream>>>(x, w3, stats, nullptr, nullptr, 0);
  fin_kernel<<<1, 32, 0, stream>>>(stats, g3, be3, 20, CNT1, S_SUM1, S_SQ1, S_SC1, S_SH1);
  conv_kernel<31, 1, 15, true><<<NWG, 32, 0, stream>>>(x, w3, stats, w13, v16, 100);

  // ---- BN13 finalize + feat ----
  fin_kernel<<<1, 32, 0, stream>>>(stats, g13, be13, 10, CNT13, S_SUM13, S_SQ13, S_SC13, S_SH13);
  feat_kernel<<<(MROWS * 160) / 256, 256, 0, stream>>>(v16, stats, feat16);

  // ---- GCN matmul ----
  gwprep_kernel<<<(112 * 160 + 255) / 256, 256, 0, stream>>>(gcn_w, gwT);
  gcn_mm_kernel<<<MROWS / 16, 32, 0, stream>>>(feat16, gwT, hbuf);

  // ---- GCN aggregation + BN2 ----
  fill_kernel<<<(NNODE + 255) / 256, 256, 0, stream>>>(deg, NNODE);
  deg_kernel<<<(EDGES + 255) / 256, 256, 0, stream>>>(eidx, edge_w, deg);
  dinv_kernel<<<(NNODE + 255) / 256, 256, 0, stream>>>(deg, dinv);
  agginit_kernel<<<(MROWS * 100) / 256, 256, 0, stream>>>(hbuf, dinv, agg);
  scatter_kernel<<<EDGES, 128, 0, stream>>>(eidx, edge_w, dinv, hbuf, agg);
  bn2_kernel<<<500, 256, 0, stream>>>(agg, bn2_g, bn2_b, sc2, sh2);
  gout_kernel<<<(MROWS * 100) / 256, 256, 0, stream>>>(agg, sc2, sh2);

  // ---- conv11 + pool -> seq ----
  zconv_kernel<<<N320, 128, 0, stream>>>(agg, w11, b11, seq);

  // ---- LSTM layer 0 (Kx=490 -> 16 k-tiles) ----
  fill_kernel<<<64, 256, 0, stream>>>(h0, BATCH * 256);
  fill_kernel<<<64, 256, 0, stream>>>(c0, BATCH * 256);
  for (int t = 0; t < WIN; ++t) {
    lstm_gates_kernel<<<256, 32, 0, stream>>>(seq + t * 490, WIN * 490, 490, 16,
                                              wih0, h0, whh0, gates);
    lstm_cell_kernel<<<64, 256, 0, stream>>>(gates, bih0, bhh0, h0, c0,
                                             out1 + t * 256, WIN * 256);
  }
  // ---- LSTM layer 1 (Kx=256 -> 8 k-tiles) ----
  fill_kernel<<<64, 256, 0, stream>>>(h1, BATCH * 256);
  fill_kernel<<<64, 256, 0, stream>>>(c1, BATCH * 256);
  for (int t = 0; t < WIN; ++t) {
    lstm_gates_kernel<<<256, 32, 0, stream>>>(out1 + t * 256, WIN * 256, 256, 8,
                                              wih1, h1, whh1, gates);
    lstm_cell_kernel<<<64, 256, 0, stream>>>(gates, bih1, bhh1, h1, c1,
                                             out2 + t * 256, WIN * 256);
  }

  // ---- attention -> output (64x256) ----
  attn_kernel<<<BATCH, 256, 0, stream>>>(out2, h1, out);
}
